// LRU_57440892617009
// MI455X (gfx1250) — matrix operand order repaired and checked by reading, not compiled, not run on hardware
//
#include <hip/hip_runtime.h>
#include <stdint.h>

typedef __attribute__((ext_vector_type(16))) _Float16 v16h;
typedef __attribute__((ext_vector_type(8)))  _Float16 v8h;
typedef __attribute__((ext_vector_type(8)))  float    v8f;

#define BBATCH 8
#define LSEQ   2048
#define HDIM   1024
#define NDIM   512
#define MROWS  (BBATCH * LSEQ)   // 16384
#define CHUNK  128
#define NCH    (LSEQ / CHUNK)    // 16

// ---------------- prep kernels ----------------

__global__ void k_prep_lam(const float* __restrict__ nu_log,
                           const float* __restrict__ theta_log,
                           float* __restrict__ lam) {
    int n = threadIdx.x;  // 512 threads
    float mod = __expf(-__expf(nu_log[n]));
    float th  = __expf(theta_log[n]);
    lam[n]        = mod * __cosf(th);
    lam[NDIM + n] = mod * __sinf(th);
}

// w1[n',h]: rows 0..511 = gamma[n]*B_re[n,h]; rows 512..1023 = gamma[n]*B_im[n,h]
__global__ void k_prep_w1(const float* __restrict__ B_re,
                          const float* __restrict__ B_im,
                          const float* __restrict__ gamma_log,
                          _Float16* __restrict__ w1) {
    int idx = blockIdx.x * 256 + threadIdx.x;
    int row = idx >> 10, h = idx & 1023;
    int n = row & (NDIM - 1);
    float g = __expf(gamma_log[n]);
    float v = (row < NDIM) ? B_re[n * HDIM + h] : B_im[n * HDIM + h];
    w1[idx] = (_Float16)(v * g);
}

// w2[o,k]: k<512 -> C_re[o,k]; 512..1023 -> -C_im[o,k-512]; >=1024 -> D[o,k-1024]
__global__ void k_prep_w2(const float* __restrict__ C_re,
                          const float* __restrict__ C_im,
                          const float* __restrict__ D,
                          _Float16* __restrict__ w2) {
    int idx = blockIdx.x * 256 + threadIdx.x;
    int o = idx >> 11, k = idx & 2047;
    float v;
    if (k < NDIM)            v =  C_re[o * NDIM + k];
    else if (k < 2 * NDIM)   v = -C_im[o * NDIM + (k - NDIM)];
    else                     v =  D[o * HDIM + (k - 2 * NDIM)];
    w2[idx] = (_Float16)v;
}

__global__ void k_f2h(const float* __restrict__ src, _Float16* __restrict__ dst) {
    int idx = blockIdx.x * 256 + threadIdx.x;
    dst[idx] = (_Float16)src[idx];
}

// ---------------- WMMA GEMM with async-DMA double buffering ----------------
// C[M, Ntot] (f32) = A'[M, K] (f16) x W[Ntot, K]^T (f16, k-contiguous rows)
// A' columns k < KS come from A0 (stride 1024), k >= KS from A1 (stride 1024).
#define BM 128
#define BN 128
#define BK 32
#define SA 40   // LDS row stride (halfwords), multiple of 8 -> aligned b128 lane reads
#define SB 40

// Per-lane async DMA: copy 16 bytes global -> LDS, tracked by ASYNCcnt.
__device__ __forceinline__ void async_copy16(uint32_t lds_addr, const void* gptr) {
    asm volatile("global_load_async_to_lds_b128 %0, %1, off"
                 :: "v"(lds_addr), "v"(gptr)
                 : "memory");
}

__global__ __launch_bounds__(256) void k_gemm(
        const _Float16* __restrict__ A0, const _Float16* __restrict__ A1, int KS,
        const _Float16* __restrict__ W, float* __restrict__ Cout,
        int Ntot, int K) {
    __shared__ _Float16 As[2][BM * SA];
    __shared__ _Float16 Bs[2][BN * SB];

    const int tid  = threadIdx.x;
    const int lane = tid & 31;
    const int wave = tid >> 5;          // 8 waves
    const int wm   = wave >> 2;         // 0..1  (64 rows each)
    const int wn   = wave & 3;          // 0..3  (32 cols each)
    const int hi   = lane >> 4;         // K-half select: k = 16*hi + j (contiguous order used here)
    const int lr   = lane & 15;

    const int row0 = blockIdx.y * BM;
    const int col0 = blockIdx.x * BN;

    const int ldr = tid >> 1;           // 0..127: tile row loaded by this thread
    const int kh  = (tid & 1) << 4;     // 0 or 16: halfword offset in k

    v8f acc[4][2] = {};

    // Issue the 4 per-thread async copies (16 halfwords of A + 16 of B) for one k-tile.
    auto issue = [&](int buf, int kt) {
        const _Float16* Abase = (kt < KS) ? (A0 + kt) : (A1 + (kt - KS));
        const _Float16* ga = Abase + (size_t)(row0 + ldr) * 1024 + kh;
        uint32_t la = (uint32_t)(uintptr_t)(&As[buf][ldr * SA + kh]);  // flat lo32 == LDS addr
        async_copy16(la,      ga);
        async_copy16(la + 16, ga + 8);
        const _Float16* gb = W + (size_t)(col0 + ldr) * K + kt + kh;
        uint32_t lb = (uint32_t)(uintptr_t)(&Bs[buf][ldr * SB + kh]);
        async_copy16(lb,      gb);
        async_copy16(lb + 16, gb + 8);
    };

    const int nk = K / BK;
    issue(0, 0);

    for (int t = 0; t < nk; ++t) {
        const int cur = t & 1;
        if (t + 1 < nk) {
            issue(cur ^ 1, (t + 1) * BK);
            // 4 newer ops in flight; async ops retire in order, so <=4 means
            // the current buffer's group has landed in LDS.
            asm volatile("s_wait_asynccnt 4" ::: "memory");
        } else {
            asm volatile("s_wait_asynccnt 0" ::: "memory");
        }
        __syncthreads();   // all threads' DMA for `cur` visible to whole WG

        // B fragments: lane holds col n = lr, halfword j -> K = 16*hi + j
        v16h bf[2];
#pragma unroll
        for (int j = 0; j < 2; ++j) {
            const _Float16* p = &Bs[cur][(wn * 32 + j * 16 + lr) * SB + hi * 16];
            v8h b0 = *(const v8h*)p;
            v8h b1 = *(const v8h*)(p + 8);
            bf[j] = __builtin_shufflevector(b0, b1, 0, 1, 2, 3, 4, 5, 6, 7,
                                            8, 9, 10, 11, 12, 13, 14, 15);
        }

#pragma unroll
        for (int i = 0; i < 4; ++i) {
            // A fragment: lane row = lr, halfword j -> K = 16*hi + j
            const _Float16* p = &As[cur][(wm * 64 + i * 16 + lr) * SA + hi * 16];
            v8h a0 = *(const v8h*)p;
            v8h a1 = *(const v8h*)(p + 8);
            v16h af = __builtin_shufflevector(a0, a1, 0, 1, 2, 3, 4, 5, 6, 7,
                                              8, 9, 10, 11, 12, 13, 14, 15);
#pragma unroll
            for (int j = 0; j < 2; ++j) {
                acc[i][j] = __builtin_amdgcn_wmma_f32_16x16x32_f16(
                    false, af, false, bf[j], (short)0, acc[i][j], false, false);
            }
        }

        __syncthreads();   // reads of `cur` done before DMA overwrites it next iter
    }

    // Epilogue: C layout — VGPR r: lane<16 -> M=r, lane>=16 -> M=r+8; N = lane%16
#pragma unroll
    for (int i = 0; i < 4; ++i) {
#pragma unroll
        for (int j = 0; j < 2; ++j) {
            int rbase = row0 + wm * 64 + i * 16 + hi * 8;
            int c     = col0 + wn * 32 + j * 16 + lr;
#pragma unroll
            for (int r = 0; r < 8; ++r)
                Cout[(size_t)(rbase + r) * Ntot + c] = acc[i][j][r];
        }
    }
}

// ---------------- chunked complex scan (3 passes) ----------------

// Pass 1: local scan within each chunk, in place in bu; record chunk-end state.
__global__ void k_scan1(float* __restrict__ bu, const float* __restrict__ lam,
                        float* __restrict__ endst) {
    int idx = blockIdx.x * 256 + threadIdx.x;   // 65536 = 8*16*512
    int n = idx & (NDIM - 1);
    int c = (idx >> 9) & (NCH - 1);
    int b = idx >> 13;
    float lre = lam[n], lim = lam[NDIM + n];
    float* p = bu + (size_t)(b * LSEQ + c * CHUNK) * 1024 + n;
    float xr = 0.f, xi = 0.f;
    for (int i = 0; i < CHUNK; ++i) {
        float br = p[0], bi = p[NDIM];
        float nr = fmaf(lre, xr, fmaf(-lim, xi, br));
        float ni = fmaf(lre, xi, fmaf( lim, xr, bi));
        xr = nr; xi = ni;
        p[0] = xr; p[NDIM] = xi;
        p += 1024;
    }
    size_t e = ((size_t)(b * NDIM + n) * NCH + c) * 2;
    endst[e] = xr; endst[e + 1] = xi;
}

// Pass 2: serial carry combine over NCH chunks; rewrite endst as carry-in per chunk.
__global__ void k_scan2(float* __restrict__ endst, const float* __restrict__ lam) {
    int idx = blockIdx.x * 256 + threadIdx.x;   // 4096 = 8*512
    int n = idx & (NDIM - 1);
    int b = idx >> 9;
    float lre = lam[n], lim = lam[NDIM + n];
    float pr = lre, pi = lim;                   // lam^CHUNK via 7 squarings (128 = 2^7)
    for (int s = 0; s < 7; ++s) {
        float t = pr * pr - pi * pi;
        pi = 2.f * pr * pi;
        pr = t;
    }
    float* e = endst + (size_t)(b * NDIM + n) * NCH * 2;
    float sr = 0.f, si = 0.f;
    for (int c = 0; c < NCH; ++c) {
        float er = e[2 * c], ei = e[2 * c + 1];
        e[2 * c] = sr; e[2 * c + 1] = si;       // carry-in for chunk c
        float nr = fmaf(pr, sr, fmaf(-pi, si, er));
        float ni = fmaf(pr, si, fmaf( pi, sr, ei));
        sr = nr; si = ni;
    }
}

// Pass 3: x_t += lam^(i+1) * carry_in; emit f16 for readout GEMM.
__global__ void k_scan3(const float* __restrict__ bu, const float* __restrict__ lam,
                        const float* __restrict__ endst, _Float16* __restrict__ xcat) {
    int idx = blockIdx.x * 256 + threadIdx.x;   // 65536
    int n = idx & (NDIM - 1);
    int c = (idx >> 9) & (NCH - 1);
    int b = idx >> 13;
    float lre = lam[n], lim = lam[NDIM + n];
    size_t eo = ((size_t)(b * NDIM + n) * NCH + c) * 2;
    float cr = endst[eo], ci = endst[eo + 1];
    const float* p = bu + (size_t)(b * LSEQ + c * CHUNK) * 1024 + n;
    _Float16* q = xcat + (size_t)(b * LSEQ + c * CHUNK) * 1024 + n;
    float pr = lre, pi = lim;
    for (int i = 0; i < CHUNK; ++i) {
        float xr = p[0]    + pr * cr - pi * ci;
        float xi = p[NDIM] + pr * ci + pi * cr;
        q[0]    = (_Float16)xr;
        q[NDIM] = (_Float16)xi;
        float t = pr * lre - pi * lim;
        pi = pr * lim + pi * lre;
        pr = t;
        p += 1024; q += 1024;
    }
}

// ---------------- launch ----------------

extern "C" void kernel_launch(void* const* d_in, const int* in_sizes, int n_in,
                              void* d_out, int out_size, void* d_ws, size_t ws_size,
                              hipStream_t stream) {
    const float* u         = (const float*)d_in[0];
    const float* nu_log    = (const float*)d_in[1];
    const float* theta_log = (const float*)d_in[2];
    const float* gamma_log = (const float*)d_in[3];
    const float* B_re      = (const float*)d_in[4];
    const float* B_im      = (const float*)d_in[5];
    const float* C_re      = (const float*)d_in[6];
    const float* C_im      = (const float*)d_in[7];
    const float* D         = (const float*)d_in[8];

    char* ws = (char*)d_ws;
    size_t off = 0;
    _Float16* u16  = (_Float16*)(ws + off); off += (size_t)MROWS * HDIM * 2;   // 32 MB
    _Float16* w1   = (_Float16*)(ws + off); off += (size_t)1024 * 1024 * 2;    //  2 MB
    _Float16* w2   = (_Float16*)(ws + off); off += (size_t)1024 * 2048 * 2;    //  4 MB
    float*    lam  = (float*)   (ws + off); off += (size_t)1024 * 4;
    float*    endc = (float*)   (ws + off); off += (size_t)BBATCH * NDIM * NCH * 2 * 4;
    float*    bu   = (float*)   (ws + off); off += (size_t)MROWS * 1024 * 4;   // 64 MB
    _Float16* xcat = (_Float16*)(ws + off); off += (size_t)MROWS * 1024 * 2;   // 32 MB

    k_prep_lam<<<1, 512, 0, stream>>>(nu_log, theta_log, lam);
    k_prep_w1<<<(1024 * 1024) / 256, 256, 0, stream>>>(B_re, B_im, gamma_log, w1);
    k_prep_w2<<<(1024 * 2048) / 256, 256, 0, stream>>>(C_re, C_im, D, w2);
    k_f2h<<<(MROWS * HDIM) / 256, 256, 0, stream>>>(u, u16);

    dim3 gg(1024 / BN, MROWS / BM);   // (8, 128)
    // GEMM1: bu[M,1024] = u16[M,1024] x w1[1024,1024]^T
    k_gemm<<<gg, 256, 0, stream>>>(u16, u16, 1024, w1, bu, 1024, 1024);

    // Chunked scan: exactly BBATCH*NCH*NDIM = 65536 threads for pass 1/3.
    k_scan1<<<(BBATCH * NCH * NDIM) / 256, 256, 0, stream>>>(bu, lam, endc);
    k_scan2<<<(BBATCH * NDIM) / 256, 256, 0, stream>>>(endc, lam);
    k_scan3<<<(BBATCH * NCH * NDIM) / 256, 256, 0, stream>>>(bu, lam, endc, xcat);

    // GEMM2: y[M,1024] = [xcat | u16][M,2048] x w2[1024,2048]^T
    k_gemm<<<gg, 256, 0, stream>>>(xcat, u16, 1024, w2, (float*)d_out, 1024, 2048);
}
